// SingleLayerMambaPGF_68539088110105
// MI455X (gfx1250) — matrix operand order repaired and checked by reading, not compiled, not run on hardware
//
#include <hip/hip_runtime.h>

typedef __attribute__((ext_vector_type(16))) _Float16 v16h;
typedef __attribute__((ext_vector_type(8)))  float    v8f;

// Problem constants (from reference setup_inputs)
constexpr int BB = 4;        // batch
constexpr int LL = 2048;     // sequence length
constexpr int DM = 256;      // model dim
constexpr int NS = 16;       // state dim
constexpr int NCHUNK = 16;   // chunks for parallel scan
constexpr int CLEN = LL / NCHUNK;  // 128

// ---------------------------------------------------------------------------
// Kernel 1: fused projection GEMM via WMMA f16 (f32 accumulate), templated on
// the output kind so the epilogue is branch-free:
//   KIND=0: dt  = softplus(u @ W_dt^T + b_dt)   [8192 x 256]  (16 col-tiles)
//   KIND=1: Bv  = u @ W_B^T + b_B               [8192 x 16]   (1 col-tile)
//   KIND=2: Cv  = u @ W_C^T + b_C               [8192 x 16]   (1 col-tile)
// One wave32 per 16x16 output tile; K fully unrolled (8 x K=32 WMMA).
// blockDim = (32, 4): 4 waves handle 4 consecutive M-tiles.
// ---------------------------------------------------------------------------
template <int KIND>
__global__ void proj_wmma_kernel(const float* __restrict__ u,
                                 const float* __restrict__ W,
                                 const float* __restrict__ bias,
                                 float* __restrict__ outp)
{
    const int lane = threadIdx.x;                    // 0..31
    const int mt   = blockIdx.x * 4 + threadIdx.y;   // M tile index 0..511
    const int jt   = blockIdx.y;                     // column tile (0..15 or 0)
    const int r    = lane & 15;
    const int half = lane >> 4;

    const float* Wslice = W + (size_t)(jt * 16) * DM;      // rows [16jt,16jt+16)
    const float* arow   = u + (size_t)(mt * 16 + r) * DM;  // A: row M = r
    const float* brow   = Wslice + (size_t)r * DM;         // B: column N = r

    v8f acc = {};
    #pragma unroll
    for (int kt = 0; kt < DM; kt += 32) {
        v16h af, bf;
        // A fragment: lane-half K pattern {2v+8h | v<4} U {16+2(v-4)+8h}
        #pragma unroll
        for (int v = 0; v < 8; ++v) {
            const int kb = (v < 4 ? 2 * v : 16 + 2 * (v - 4)) + 8 * half + kt;
            const float2 av = *(const float2*)(arow + kb);
            af[2 * v]     = (_Float16)av.x;
            af[2 * v + 1] = (_Float16)av.y;
        }
        // B fragment: lane-half K pattern {2v+8h | v<4} U {16+2(v-4)+8h}
        #pragma unroll
        for (int v = 0; v < 8; ++v) {
            const int kb = (v < 4 ? 2 * v : 16 + 2 * (v - 4)) + 8 * half + kt;
            const float2 bv = *(const float2*)(brow + kb);
            bf[2 * v]     = (_Float16)bv.x;
            bf[2 * v + 1] = (_Float16)bv.y;
        }
        acc = __builtin_amdgcn_wmma_f32_16x16x32_f16(
            /*neg_a=*/false, af, /*neg_b=*/false, bf,
            /*c_mod=*/(short)0, acc, /*reuse_a=*/false, /*reuse_b=*/false);
    }

    // Epilogue: C/D layout -> VGPR j holds row M = mt*16 + 8*half + j, col N = r
    const float bia = bias[jt * 16 + r];
    #pragma unroll
    for (int j = 0; j < 8; ++j) {
        const int m = mt * 16 + 8 * half + j;
        const float val = acc[j] + bia;
        if constexpr (KIND == 0) {
            const float sp = (val > 20.0f) ? val : log1pf(__expf(val));  // softplus
            outp[(size_t)m * DM + jt * 16 + r] = sp;
        } else {
            outp[(size_t)m * NS + r] = val;
        }
    }
}

// ---------------------------------------------------------------------------
// Shared per-step math: dA = exp(dt*A), bb = phi1(dt*A) * B_vec * u
// Single transcendental: em = expm1(z);  dA = em + 1;  phi1 = em / z.
// ---------------------------------------------------------------------------
__device__ __forceinline__ void step_coeffs(float dt, float uv, float Bv, float An,
                                            float& dA, float& bb)
{
    const float z  = dt * An;
    const float em = expm1f(z);
    const float phi = (fabsf(z) < 1e-12f) ? 1.0f : em / z;
    dA = em + 1.0f;
    bb = phi * Bv * uv;
}

// ---------------------------------------------------------------------------
// Phase A: per-chunk carries.  For each state (b, dm, n) and chunk c:
//   carryA = prod(dA_t),  carryB = h at chunk end assuming h_in = 0.
// block = 256 threads = 16 dm x 16 n;  grid = (NCHUNK, DM/16, BB)
// ---------------------------------------------------------------------------
__global__ void scan_phaseA(const float* __restrict__ dt, const float* __restrict__ u,
                            const float* __restrict__ Bv, const float* __restrict__ A_log,
                            float* __restrict__ carryA, float* __restrict__ carryB)
{
    const int n   = threadIdx.x & 15;
    const int dml = threadIdx.x >> 4;
    const int c   = blockIdx.x;
    const int dm  = blockIdx.y * 16 + dml;
    const int b   = blockIdx.z;
    const float An = -__expf(A_log[n]);

    const int t0 = c * CLEN;
    const float* dtp = dt + ((size_t)b * LL + t0) * DM + dm;
    const float* up  = u  + ((size_t)b * LL + t0) * DM + dm;
    const float* bp  = Bv + ((size_t)b * LL + t0) * NS + n;

    float aprod = 1.0f, h = 0.0f;
    for (int t = 0; t < CLEN; ++t) {
        float dA, bb;
        step_coeffs(dtp[(size_t)t * DM], up[(size_t)t * DM], bp[(size_t)t * NS], An, dA, bb);
        h = fmaf(dA, h, bb);
        aprod *= dA;
    }
    const size_t sidx = (((size_t)b * DM + dm) * NS + n) * NCHUNK + c;
    carryA[sidx] = aprod;
    carryB[sidx] = h;
}

// ---------------------------------------------------------------------------
// Phase B: sequential combine of chunk carries -> initial h for each chunk.
// block = 256 threads;  grid = (DM/16, BB)
// ---------------------------------------------------------------------------
__global__ void scan_phaseB(const float* __restrict__ carryA,
                            const float* __restrict__ carryB,
                            float* __restrict__ hinit)
{
    const int n   = threadIdx.x & 15;
    const int dml = threadIdx.x >> 4;
    const int dm  = blockIdx.x * 16 + dml;
    const int b   = blockIdx.y;
    const size_t s = (((size_t)b * DM + dm) * NS + n) * NCHUNK;

    float h = 0.0f;
    #pragma unroll
    for (int c = 0; c < NCHUNK; ++c) {
        hinit[s + c] = h;
        h = fmaf(carryA[s + c], h, carryB[s + c]);
    }
}

// ---------------------------------------------------------------------------
// Phase C: replay each chunk from its true initial state, fuse the einsum
// over n (16-wide shfl_xor butterfly inside the wave), + D*u, * scale.
// block = 256 threads = 16 dm x 16 n;  grid = (NCHUNK, DM/16, BB)
// ---------------------------------------------------------------------------
__global__ void scan_phaseC(const float* __restrict__ dt, const float* __restrict__ u,
                            const float* __restrict__ Bv, const float* __restrict__ Cv,
                            const float* __restrict__ A_log,
                            const float* __restrict__ Dp, const float* __restrict__ scl,
                            const float* __restrict__ hinit,
                            float* __restrict__ out)
{
    const int n   = threadIdx.x & 15;
    const int dml = threadIdx.x >> 4;
    const int c   = blockIdx.x;
    const int dm  = blockIdx.y * 16 + dml;
    const int b   = blockIdx.z;
    const float An = -__expf(A_log[n]);
    const float Dd = Dp[dm];
    const float sc = scl[dm];

    const int t0 = c * CLEN;
    const float* dtp = dt + ((size_t)b * LL + t0) * DM + dm;
    const float* up  = u  + ((size_t)b * LL + t0) * DM + dm;
    const float* bp  = Bv + ((size_t)b * LL + t0) * NS + n;
    const float* cp  = Cv + ((size_t)b * LL + t0) * NS + n;
    float* op = out + ((size_t)b * LL + t0) * DM + dm;

    float h = hinit[(((size_t)b * DM + dm) * NS + n) * NCHUNK + c];

    for (int t = 0; t < CLEN; ++t) {
        const float uv = up[(size_t)t * DM];
        float dA, bb;
        step_coeffs(dtp[(size_t)t * DM], uv, bp[(size_t)t * NS], An, dA, bb);
        h = fmaf(dA, h, bb);
        float contrib = cp[(size_t)t * NS] * h;
        // reduce over the 16 n-lanes (lanes 0-15 and 16-31 are distinct dm)
        contrib += __shfl_xor(contrib, 8, 16);
        contrib += __shfl_xor(contrib, 4, 16);
        contrib += __shfl_xor(contrib, 2, 16);
        contrib += __shfl_xor(contrib, 1, 16);
        if (n == 0) {
            op[(size_t)t * DM] = (contrib + Dd * uv) * sc;
        }
    }
}

// ---------------------------------------------------------------------------
extern "C" void kernel_launch(void* const* d_in, const int* in_sizes, int n_in,
                              void* d_out, int out_size, void* d_ws, size_t ws_size,
                              hipStream_t stream)
{
    const float* u      = (const float*)d_in[0];
    const float* A_log  = (const float*)d_in[1];
    const float* D_par  = (const float*)d_in[2];
    const float* scale  = (const float*)d_in[3];
    const float* W_dt   = (const float*)d_in[4];
    const float* b_dt   = (const float*)d_in[5];
    const float* W_B    = (const float*)d_in[6];
    const float* b_B    = (const float*)d_in[7];
    const float* W_C    = (const float*)d_in[8];
    const float* b_C    = (const float*)d_in[9];
    float* out = (float*)d_out;

    // Workspace layout (floats), everything written before read each call
    float* ws     = (float*)d_ws;
    float* dtbuf  = ws;                                     // 8192*256 = 2,097,152
    float* Bvbuf  = dtbuf + (size_t)BB * LL * DM;           // 8192*16  =   131,072
    float* Cvbuf  = Bvbuf + (size_t)BB * LL * NS;           //             131,072
    float* carryA = Cvbuf + (size_t)BB * LL * NS;           // 4*256*16*16 = 262,144
    float* carryB = carryA + (size_t)BB * DM * NS * NCHUNK;
    float* hinit  = carryB + (size_t)BB * DM * NS * NCHUNK;

    // 1) Projections (WMMA): dt (256 cols), Bv (16 cols), Cv (16 cols)
    {
        dim3 block(32, 4);                       // 4 waves per block
        dim3 grid_dt((BB * LL / 16) / 4, 16);    // (128, 16) column tiles of dt
        proj_wmma_kernel<0><<<grid_dt, block, 0, stream>>>(u, W_dt, b_dt, dtbuf);

        dim3 grid_bc((BB * LL / 16) / 4, 1);     // (128, 1) single 16-col tile
        proj_wmma_kernel<1><<<grid_bc, block, 0, stream>>>(u, W_B, b_B, Bvbuf);
        proj_wmma_kernel<2><<<grid_bc, block, 0, stream>>>(u, W_C, b_C, Cvbuf);
    }

    // 2) Chunk-parallel linear-recurrence scan
    {
        dim3 gridA(NCHUNK, DM / 16, BB);     // (16, 16, 4) = 1024 blocks
        scan_phaseA<<<gridA, 256, 0, stream>>>(dtbuf, u, Bvbuf, A_log, carryA, carryB);

        dim3 gridB(DM / 16, BB);             // (16, 4)
        scan_phaseB<<<gridB, 256, 0, stream>>>(carryA, carryB, hinit);

        scan_phaseC<<<gridA, 256, 0, stream>>>(dtbuf, u, Bvbuf, Cvbuf, A_log,
                                               D_par, scale, hinit, out);
    }
}
